// Quantization_Head_38577396252890
// MI455X (gfx1250) — matrix operand order repaired and checked by reading, not compiled, not run on hardware
//
#include <hip/hip_runtime.h>

// ---------------------------------------------------------------------------
// Fused quantization head for MI455X (gfx1250, wave32, WMMA).
//   prepass:  Wbf = bf16(W), Cbf = bf16(C), CbfT[d][m] = bf16(C[m][d]),
//             c2all[book][m] = ||C[m, book*64:(book+1)*64]||^2   (all in d_ws)
//   main:     X = x @ W^T + b                      (bf16 WMMA, fp32 accum)
//             logits = 2*tau*(Xb.Cb) - tau*|c|^2   (x^2 row-term cancels)
//             soft = softmax(logits); Z = soft @ Cb
// One wave32 owns a 16-row x 1-book tile; 2 waves/WG; grid ordered so the
// 8 book-WGs of a row tile are adjacent (x rows stay hot in L2).
// ---------------------------------------------------------------------------

typedef __attribute__((ext_vector_type(16))) __bf16 v16bf;
typedef __attribute__((ext_vector_type(8)))  __bf16 v8bf;
typedef __attribute__((ext_vector_type(2)))  __bf16 v2bf;
typedef __attribute__((ext_vector_type(8)))  float  v8f;

#define TAU_Q 5.0f
constexpr int NROWS = 65536;
constexpr int DIN   = 512;
constexpr int DOUT  = 512;
constexpr int NWRD  = 256;   // codewords per book
constexpr int LB    = 64;    // dims per book
constexpr int NBOOK = 8;

// workspace layout (bytes)
constexpr size_t WS_WBF  = 0;                                   // 512 KB
constexpr size_t WS_CBF  = WS_WBF + (size_t)DOUT * DIN * 2;     // 256 KB
constexpr size_t WS_CBFT = WS_CBF + (size_t)NWRD * DOUT * 2;    // 256 KB
constexpr size_t WS_C2   = WS_CBFT + (size_t)DOUT * NWRD * 2;   //   8 KB

__device__ __forceinline__ void pack8(v16bf& v, int base, float4 lo, float4 hi) {
    v[base+0] = (__bf16)lo.x; v[base+1] = (__bf16)lo.y;
    v[base+2] = (__bf16)lo.z; v[base+3] = (__bf16)lo.w;
    v[base+4] = (__bf16)hi.x; v[base+5] = (__bf16)hi.y;
    v[base+6] = (__bf16)hi.z; v[base+7] = (__bf16)hi.w;
}
__device__ __forceinline__ void put8(v16bf& v, int base, v8bf s) {
    #pragma unroll
    for (int i = 0; i < 8; ++i) v[base + i] = s[i];
}

// ---- prepass: convert W/C to bf16, build transposed codebook copy ----
__global__ __launch_bounds__(256)
void cvt_wc(const float* __restrict__ W, const float* __restrict__ C,
            __bf16* __restrict__ Wbf, __bf16* __restrict__ Cbf,
            __bf16* __restrict__ CbfT)
{
    int i = blockIdx.x * 256 + threadIdx.x;          // 0 .. 262143
    Wbf[i] = (__bf16)W[i];
    if (i < NWRD * DOUT) {
        int m = i >> 9, d = i & 511;
        float c = C[i];
        Cbf[i] = (__bf16)c;
        CbfT[(size_t)d * NWRD + m] = (__bf16)c;      // [d][m]
    }
}

// ---- prepass: per-book codeword squared norms ----
__global__ __launch_bounds__(256)
void c2_kernel(const float* __restrict__ C, float* __restrict__ c2all)
{
    int i = blockIdx.x * 256 + threadIdx.x;          // 0 .. 2047
    int book = i >> 8, m = i & 255;
    const float* cr = C + (size_t)m * DOUT + book * LB;
    float s = 0.f;
    #pragma unroll 8
    for (int l = 0; l < LB; ++l) s += cr[l] * cr[l];
    c2all[i] = s;
}

__global__ __launch_bounds__(64)
void pq_head_fused(const float* __restrict__ x,
                   const __bf16* __restrict__ Wbf,
                   const float* __restrict__ bias,
                   const __bf16* __restrict__ Cbf,
                   const __bf16* __restrict__ CbfT,
                   const float* __restrict__ c2all,
                   float* __restrict__ Xout,
                   float* __restrict__ Zout)
{
    __shared__ __bf16 cbsT[LB][NWRD + 8];   // codebook^T, padded rows (33 KB)
    __shared__ float  c2s[NWRD];            // ||c||^2 per codeword    (1 KB)
    __shared__ __bf16 xs[2][16][LB];        // per-wave X tile         (4 KB)
    __shared__ __bf16 softs[2][16][NWRD];   // per-wave softmax       (16 KB)

    const int tid  = threadIdx.x;
    const int lane = tid & 31;
    const int wave = tid >> 5;
    const int half = lane >> 4;     // 0: lanes 0-15, 1: lanes 16-31
    const int ln16 = lane & 15;
    const int book = blockIdx.x & 7;          // books adjacent -> x hot in L2
    const int cb0  = book * LB;
    const int r0   = (blockIdx.x >> 3) * 32 + wave * 16;

    // ---- preload transposed codebook slice (uint4 copies) + c2 ----
    for (int idx = tid; idx < (NWRD * LB) / 8; idx += 64) {   // 2048 chunks
        int l  = idx >> 5;               // 32 chunks of 8 per 256-row
        int m8 = (idx & 31) * 8;
        uint4 v = *(const uint4*)(CbfT + (size_t)(cb0 + l) * NWRD + m8);
        *(uint4*)&cbsT[l][m8] = v;
    }
    for (int m = tid; m < NWRD; m += 64) c2s[m] = c2all[book * NWRD + m];
    __syncthreads();

    // =============== stage 1: X_tile[16x64] = x @ W^T + b ===============
    v8f acc[4];
    #pragma unroll
    for (int t = 0; t < 4; ++t)
        #pragma unroll
        for (int j = 0; j < 8; ++j) acc[t][j] = 0.f;

    #pragma unroll 1
    for (int k0 = 0; k0 < DIN; k0 += 32) {
        // A fragment (fp32 x, converted in-flight): row = ln16,
        // K contiguous {16h..16h+15}
        const float* xr = x + (size_t)(r0 + ln16) * DIN + k0 + half * 16;
        v16bf a;
        pack8(a, 0, *(const float4*)(xr + 0),  *(const float4*)(xr + 4));
        pack8(a, 8, *(const float4*)(xr + 8), *(const float4*)(xr + 12));

        #pragma unroll
        for (int t = 0; t < 4; ++t) {
            // B[k][n] = Wbf[cb0 + t*16 + n][k]; 16 contiguous bf16 per lane
            const __bf16* wr = Wbf + (size_t)(cb0 + t * 16 + ln16) * DIN
                                   + k0 + half * 16;
            v16bf bb;
            put8(bb, 0, *(const v8bf*)(wr + 0));
            put8(bb, 8, *(const v8bf*)(wr + 8));
            acc[t] = __builtin_amdgcn_wmma_f32_16x16x32_bf16(
                false, a, false, bb, (short)0, acc[t], false, false);
        }
    }

    // bias add, store X, stage bf16 copy for stage-2 A operand
    #pragma unroll
    for (int t = 0; t < 4; ++t) {
        float bv = bias[cb0 + t * 16 + ln16];
        #pragma unroll
        for (int j = 0; j < 8; ++j) {
            float v = acc[t][j] + bv;
            int row = j + 8 * half;
            Xout[(size_t)(r0 + row) * DOUT + cb0 + t * 16 + ln16] = v;
            xs[wave][row][t * 16 + ln16] = (__bf16)v;
        }
    }
    __syncthreads();

    // ====== stage 2: xc[16x256] = Xb . Cb^T  (16 N-tiles, K=64) ======
    v8f xc[16];
    #pragma unroll
    for (int t = 0; t < 16; ++t)
        #pragma unroll
        for (int j = 0; j < 8; ++j) xc[t][j] = 0.f;

    v16bf a2[2];
    #pragma unroll
    for (int kk = 0; kk < 2; ++kk)
        #pragma unroll
        for (int e = 0; e < 16; e += 2) {
            int kl = 16 * half + e;
            v2bf p = *(const v2bf*)&xs[wave][ln16][kk * 32 + kl];
            a2[kk][e] = p[0]; a2[kk][e + 1] = p[1];
        }

    #pragma unroll
    for (int t = 0; t < 16; ++t)
        #pragma unroll
        for (int kk = 0; kk < 2; ++kk) {
            // B[k=l][n=m] = Cbf[m][cb0+l]; contiguous 16 bf16 from L2
            const __bf16* cr = Cbf + (size_t)(t * 16 + ln16) * DOUT
                                   + cb0 + kk * 32 + half * 16;
            v16bf bb;
            put8(bb, 0, *(const v8bf*)(cr + 0));
            put8(bb, 8, *(const v8bf*)(cr + 8));
            xc[t] = __builtin_amdgcn_wmma_f32_16x16x32_bf16(
                false, a2[kk], false, bb, (short)0, xc[t], false, false);
        }

    // logits = 2*tau*xc - tau*|c|^2  (row-constant -tau*|x|^2 cancels)
    #pragma unroll
    for (int t = 0; t < 16; ++t) {
        float c2v = c2s[t * 16 + ln16];
        #pragma unroll
        for (int j = 0; j < 8; ++j)
            xc[t][j] = 2.f * TAU_Q * xc[t][j] - TAU_Q * c2v;
    }

    // softmax over 256 codewords: per row, values span 16 lanes x 16 tiles
    #pragma unroll
    for (int j = 0; j < 8; ++j) {
        float mx = xc[0][j];
        #pragma unroll
        for (int t = 1; t < 16; ++t) mx = fmaxf(mx, xc[t][j]);
        #pragma unroll
        for (int off = 1; off < 16; off <<= 1)
            mx = fmaxf(mx, __shfl_xor(mx, off, 16));
        float s = 0.f;
        #pragma unroll
        for (int t = 0; t < 16; ++t) {
            float e = __expf(xc[t][j] - mx);
            xc[t][j] = e;
            s += e;
        }
        #pragma unroll
        for (int off = 1; off < 16; off <<= 1)
            s += __shfl_xor(s, off, 16);
        float inv = 1.f / s;
        #pragma unroll
        for (int t = 0; t < 16; ++t) xc[t][j] *= inv;
    }

    // stage softmax weights (bf16) for stage-3 A operand
    #pragma unroll
    for (int t = 0; t < 16; ++t)
        #pragma unroll
        for (int j = 0; j < 8; ++j)
            softs[wave][j + 8 * half][t * 16 + ln16] = (__bf16)xc[t][j];
    __syncthreads();

    // ====== stage 3: Z[16x64] = soft[16x256] @ Cb[256x64] ======
    v8f zac[4];
    #pragma unroll
    for (int t = 0; t < 4; ++t)
        #pragma unroll
        for (int j = 0; j < 8; ++j) zac[t][j] = 0.f;

    #pragma unroll 1
    for (int k0 = 0; k0 < NWRD; k0 += 32) {
        v16bf a3;
        #pragma unroll
        for (int e = 0; e < 16; e += 2) {
            int kl = 16 * half + e;
            v2bf p = *(const v2bf*)&softs[wave][ln16][k0 + kl];
            a3[e] = p[0]; a3[e + 1] = p[1];
        }
        #pragma unroll
        for (int t = 0; t < 4; ++t) {
            // B[k=m][n=l] = cbsT[l][m]; contiguous dword pairs along m
            v16bf bb;
            #pragma unroll
            for (int e = 0; e < 16; e += 2) {
                v2bf p = *(const v2bf*)&cbsT[t * 16 + ln16][k0 + 16 * half + e];
                bb[e] = p[0]; bb[e + 1] = p[1];
            }
            zac[t] = __builtin_amdgcn_wmma_f32_16x16x32_bf16(
                false, a3, false, bb, (short)0, zac[t], false, false);
        }
    }

    #pragma unroll
    for (int t = 0; t < 4; ++t)
        #pragma unroll
        for (int j = 0; j < 8; ++j)
            Zout[(size_t)(r0 + j + 8 * half) * DOUT + cb0 + t * 16 + ln16] = zac[t][j];
}

extern "C" void kernel_launch(void* const* d_in, const int* in_sizes, int n_in,
                              void* d_out, int out_size, void* d_ws, size_t ws_size,
                              hipStream_t stream) {
    const float* x = (const float*)d_in[0];   // [65536, 512]
    const float* W = (const float*)d_in[1];   // [512, 512]
    const float* b = (const float*)d_in[2];   // [512]
    const float* C = (const float*)d_in[3];   // [256, 512]
    float* Xout = (float*)d_out;                       // [65536, 512]
    float* Zout = Xout + (size_t)NROWS * DOUT;         // [65536, 512]

    char* ws = (char*)d_ws;
    __bf16* Wbf   = (__bf16*)(ws + WS_WBF);
    __bf16* Cbf   = (__bf16*)(ws + WS_CBF);
    __bf16* CbfT  = (__bf16*)(ws + WS_CBFT);
    float*  c2all = (float*)(ws + WS_C2);

    cvt_wc<<<dim3((DOUT * DIN) / 256), dim3(256), 0, stream>>>(W, C, Wbf, Cbf, CbfT);
    c2_kernel<<<dim3((NBOOK * NWRD) / 256), dim3(256), 0, stream>>>(C, c2all);

    // 8 book-WGs of one 32-row tile are adjacent in the grid
    dim3 grid((NROWS / 32) * NBOOK);
    pq_head_fused<<<grid, dim3(64), 0, stream>>>(x, Wbf, b, Cbf, CbfT, c2all,
                                                 Xout, Zout);
}
